// MultiHeadAttentionBlock_14783277432888
// MI455X (gfx1250) — matrix operand order repaired and checked by reading, not compiled, not run on hardware
//
#include <hip/hip_runtime.h>
#include <hip/hip_bf16.h>
#include <math.h>

#define Bsz 2
#define Ssz 2048
#define Esz 1024
#define Hsz 16
#define DKsz 64

typedef __attribute__((ext_vector_type(16))) __bf16 v16bf;
typedef __attribute__((ext_vector_type(8)))  __bf16 v8bf;
typedef __attribute__((ext_vector_type(4)))  __bf16 v4bf;
typedef __attribute__((ext_vector_type(8)))  float  v8f;

// GCC-style int4 vector, matching the async-LDS builtin prototype
// (param prints as '__attribute__((__vector_size__(4 * sizeof(int)))) int __device__ *').
typedef int v4i_gcc __attribute__((vector_size(16)));
typedef __attribute__((address_space(1))) v4i_gcc* global_v4i_ptr;
typedef __attribute__((address_space(3))) v4i_gcc* lds_v4i_ptr;

#if defined(__has_builtin)
#if __has_builtin(__builtin_amdgcn_global_load_async_to_lds_b128)
#define HAVE_ASYNC_LDS 1
#endif
#endif

#ifndef HAVE_ASYNC_LDS
#define HAVE_ASYNC_LDS 0
#endif

static __device__ __forceinline__ void wait_asynccnt0() {
#if defined(__has_builtin) && __has_builtin(__builtin_amdgcn_s_wait_asynccnt)
  __builtin_amdgcn_s_wait_asynccnt(0);
#else
  asm volatile("s_wait_asynccnt 0" ::: "memory");
#endif
}

static __device__ __forceinline__ v8f zero_v8f() {
  v8f z;
#pragma unroll
  for (int i = 0; i < 8; ++i) z[i] = 0.0f;
  return z;
}

// Build a 16-wide bf16 fragment from two 16-byte chunks (two ds/global b128 loads).
static __device__ __forceinline__ v16bf make_frag(const __bf16* p0, const __bf16* p1) {
  v8bf lo = *(const v8bf*)p0;
  v8bf hi = *(const v8bf*)p1;
  v16bf r;
#pragma unroll
  for (int i = 0; i < 8; ++i) { r[i] = lo[i]; r[i + 8] = hi[i]; }
  return r;
}

// C = A[M,K] @ W[N,K]^T + bias, bf16 WMMA with f32 accumulate.
// Block tile 128x128, K-step 32, 8 waves; wave w computes rows [16w,16w+16) x all 128 cols.
// OUT_MODE: 0 = f32 row-major [M,N]; 1 = bf16 row-major [M,N];
//           2 = bf16 per-head transposed [B,H,DK,S] (token contiguous -> b128 stores).
template <typename TIn, int OUT_MODE>
__global__ __launch_bounds__(256) void gemm_bias_wmma(
    const TIn* __restrict__ A, const float* __restrict__ W,
    const float* __restrict__ bias, void* __restrict__ Cout,
    int M, int N, int K)
{
  __shared__ __bf16 As[128 * 32];
  __shared__ __bf16 Ws[128 * 32];

  const int t    = threadIdx.x;
  const int wave = t >> 5;
  const int lane = t & 31;
  const int hi16 = lane >> 4;   // which 16-lane half of the wave
  const int l16  = lane & 15;
  const int m0   = blockIdx.y * 128;
  const int n0   = blockIdx.x * 128;

  v8f acc[8];
#pragma unroll
  for (int i = 0; i < 8; ++i) acc[i] = zero_v8f();

  for (int k0 = 0; k0 < K; k0 += 32) {
    // ---- stage 128x32 A tile and 128x32 W tile to LDS as bf16 ----
#pragma unroll
    for (int i = 0; i < 4; ++i) {
      int c   = t + i * 256;        // 0..1023 chunks of 4 elements
      int row = c >> 3;
      int col = (c & 7) << 2;
      {
        const TIn* src = A + (size_t)(m0 + row) * K + (k0 + col);
        v4bf d;
        if constexpr (sizeof(TIn) == 4) {
          float4 f = *(const float4*)src;
          d[0] = (__bf16)f.x; d[1] = (__bf16)f.y; d[2] = (__bf16)f.z; d[3] = (__bf16)f.w;
        } else {
          d = *(const v4bf*)src;
        }
        *(v4bf*)&As[row * 32 + col] = d;
      }
      {
        const float* src = W + (size_t)(n0 + row) * K + (k0 + col);
        float4 f = *(const float4*)src;
        v4bf d;
        d[0] = (__bf16)f.x; d[1] = (__bf16)f.y; d[2] = (__bf16)f.z; d[3] = (__bf16)f.w;
        *(v4bf*)&Ws[row * 32 + col] = d;
      }
    }
    // hint next A tile into cache while we compute
    if (k0 + 32 < K)
      __builtin_prefetch(A + (size_t)(m0 + (t >> 1)) * K + (k0 + 32), 0, 1);
    __syncthreads();

    // A fragment: lane holds M=l16; K halves {0..7,16..23} (lanes 0-15) / {8..15,24..31}
    const __bf16* ap = &As[(wave * 16 + l16) * 32 + hi16 * 8];
    v16bf afrag = make_frag(ap, ap + 16);
#pragma unroll
    for (int nt = 0; nt < 8; ++nt) {
      // B fragment: lane holds N=l16 of tile nt; K halves {0..7,16..23} (lanes 0-15) / {8..15,24..31}
      const __bf16* bp = &Ws[(nt * 16 + l16) * 32 + hi16 * 8];
      v16bf bfrag = make_frag(bp, bp + 16);
      acc[nt] = __builtin_amdgcn_wmma_f32_16x16x32_bf16(
          false, afrag, false, bfrag, (short)0, acc[nt], false, false);
    }
    __syncthreads();
  }

  // ---- epilogue: add bias, store ----
#pragma unroll
  for (int nt = 0; nt < 8; ++nt) {
    int n    = n0 + nt * 16 + l16;
    float bv = bias[n];
    if constexpr (OUT_MODE == 2) {
      // rows per lane are 8 consecutive tokens -> one 16B store into [B,H,DK,S]
      int m_base = m0 + wave * 16 + hi16 * 8;       // multiple of 8
      int bb  = m_base / Ssz;
      int tok = m_base % Ssz;
      int hh  = n >> 6;         // n / DK
      int dd  = n & 63;         // n % DK
      __bf16* dst = (__bf16*)Cout +
          (((size_t)bb * Hsz + hh) * DKsz + dd) * Ssz + tok;
      v8bf pk;
#pragma unroll
      for (int r = 0; r < 8; ++r) pk[r] = (__bf16)(acc[nt][r] + bv);
      *(v8bf*)dst = pk;
    } else {
#pragma unroll
      for (int r = 0; r < 8; ++r) {
        int m     = m0 + wave * 16 + r + hi16 * 8;
        float val = acc[nt][r] + bv;
        if constexpr (OUT_MODE == 1)
          ((__bf16*)Cout)[(size_t)m * N + n] = (__bf16)val;
        else
          ((float*)Cout)[(size_t)m * N + n] = val;
      }
    }
  }
}

// Flash attention, causal. Grid: (S/128, B*H). 8 waves/block, 16 query rows per wave.
// Qp/Kp: bf16 [B,S,E] (head h at column h*DK). Vt: bf16 [B,H,DK,S] (pre-transposed).
__global__ __launch_bounds__(256) void flash_attn_wmma(
    const __bf16* __restrict__ Qp, const __bf16* __restrict__ Kp,
    const __bf16* __restrict__ Vt, __bf16* __restrict__ ctx)
{
  __shared__ __bf16 Ks[32 * 64];      // [key][d]
  __shared__ __bf16 Vs[64 * 32];      // [d][key]  (already transposed in global)
  __shared__ __bf16 Ps[8][16 * 32];   // per-wave probability tile

  const int t    = threadIdx.x;
  const int wave = t >> 5;
  const int lane = t & 31;
  const int hi16 = lane >> 4;
  const int l16  = lane & 15;
  const int bh   = blockIdx.y;
  const int b    = bh / Hsz;
  const int h    = bh % Hsz;
  const int q0b  = blockIdx.x * 128;
  const int q0w  = q0b + wave * 16;
  const size_t base  = ((size_t)b * Ssz) * Esz + (size_t)h * DKsz;          // [B,S,E] head base
  const size_t vbase = ((size_t)b * Hsz + h) * DKsz * Ssz;                  // [B,H,DK,S] head base

  // Q fragments held in registers across the whole kv loop (2 K-steps over DK=64)
  v16bf qf[2];
  {
    const __bf16* qrow = Qp + base + (size_t)(q0w + l16) * Esz;
#pragma unroll
    for (int ks = 0; ks < 2; ++ks) {
      const __bf16* p = qrow + ks * 32 + hi16 * 8;
      qf[ks] = make_frag(p, p + 16);
    }
  }

  v8f o[4];
#pragma unroll
  for (int i = 0; i < 4; ++i) o[i] = zero_v8f();
  float mrun[8], lrun[8];
#pragma unroll
  for (int r = 0; r < 8; ++r) { mrun[r] = -1e30f; lrun[r] = 0.0f; }

  const int ktmax = (q0b + 127) >> 5;   // inclusive last 32-key tile for this block
  for (int kt = 0; kt <= ktmax; ++kt) {
    // ---- cooperative K/V tile staging: one 16B chunk each per thread ----
    {
      int key = t >> 3;                 // K tile: 32 keys x 64 d, 8 chunks/row
      int d0  = (t & 7) * 8;
      const __bf16* ksrc = Kp + base + (size_t)(kt * 32 + key) * Esz + d0;
      int dr   = t >> 2;                // V tile: 64 d-rows x 32 keys, 4 chunks/row
      int key0 = (t & 3) * 8;
      const __bf16* vsrc = Vt + vbase + (size_t)dr * Ssz + kt * 32 + key0;
#if HAVE_ASYNC_LDS
      // CDNA5 async memory->LDS copy (ASYNCcnt), no VGPR round trip.
      __builtin_amdgcn_global_load_async_to_lds_b128(
          (global_v4i_ptr)(void*)ksrc,
          (lds_v4i_ptr)&Ks[key * 64 + d0], 0, 0);
      __builtin_amdgcn_global_load_async_to_lds_b128(
          (global_v4i_ptr)(void*)vsrc,
          (lds_v4i_ptr)&Vs[dr * 32 + key0], 0, 0);
      wait_asynccnt0();
#else
      *(v8bf*)&Ks[key * 64 + d0]  = *(const v8bf*)ksrc;
      *(v8bf*)&Vs[dr * 32 + key0] = *(const v8bf*)vsrc;
#endif
    }
    __syncthreads();

    if (kt * 32 <= q0w + 15) {
      // ---- scores: S = (Q @ K^T) * 1/sqrt(DK), two 16-key N tiles ----
      v8f s[2];
      s[0] = zero_v8f(); s[1] = zero_v8f();
#pragma unroll
      for (int ks = 0; ks < 2; ++ks) {
#pragma unroll
        for (int nt = 0; nt < 2; ++nt) {
          const __bf16* bp = &Ks[(nt * 16 + l16) * 64 + ks * 32 + hi16 * 8];
          v16bf bf = make_frag(bp, bp + 16);
          s[nt] = __builtin_amdgcn_wmma_f32_16x16x32_bf16(
              false, qf[ks], false, bf, (short)0, s[nt], false, false);
        }
      }
      // ---- causal mask + online softmax (row spread across 16 lanes) ----
#pragma unroll
      for (int r = 0; r < 8; ++r) {
        int qg  = q0w + r + hi16 * 8;
        int kg0 = kt * 32 + l16;
        float v0 = s[0][r] * 0.125f;
        float v1 = s[1][r] * 0.125f;
        if (kg0 > qg)      v0 = -1e30f;
        if (kg0 + 16 > qg) v1 = -1e30f;
        float mx = fmaxf(v0, v1);
#pragma unroll
        for (int off = 1; off < 16; off <<= 1)
          mx = fmaxf(mx, __shfl_xor(mx, off, 16));
        float mnew  = fmaxf(mrun[r], mx);
        float alpha = __expf(mrun[r] - mnew);
        float p0 = __expf(v0 - mnew);
        float p1 = __expf(v1 - mnew);
        float rs = p0 + p1;
#pragma unroll
        for (int off = 1; off < 16; off <<= 1)
          rs += __shfl_xor(rs, off, 16);
        lrun[r] = lrun[r] * alpha + rs;
        mrun[r] = mnew;
#pragma unroll
        for (int j = 0; j < 4; ++j) o[j][r] *= alpha;
        int prow = r + hi16 * 8;
        Ps[wave][prow * 32 + l16]      = (__bf16)p0;
        Ps[wave][prow * 32 + 16 + l16] = (__bf16)p1;
      }
      // DS ops from one wave are in-order, but make the dependency explicit.
      asm volatile("s_wait_dscnt 0" ::: "memory");

      // ---- O += P @ V  (A = P 16x32, B = V^T rows, 4 dk tiles) ----
      const __bf16* pp = &Ps[wave][l16 * 32 + hi16 * 8];
      v16bf pf = make_frag(pp, pp + 16);
#pragma unroll
      for (int j = 0; j < 4; ++j) {
        const __bf16* bp = &Vs[(j * 16 + l16) * 32 + hi16 * 8];
        v16bf bf = make_frag(bp, bp + 16);
        o[j] = __builtin_amdgcn_wmma_f32_16x16x32_bf16(
            false, pf, false, bf, (short)0, o[j], false, false);
      }
    }
    __syncthreads();
  }

  // ---- normalize and write context as bf16 [B,S,E] ----
#pragma unroll
  for (int j = 0; j < 4; ++j) {
    int d = j * 16 + l16;
#pragma unroll
    for (int r = 0; r < 8; ++r) {
      int qg    = q0w + r + hi16 * 8;
      float val = o[j][r] / lrun[r];
      ctx[base + (size_t)qg * Esz + d] = (__bf16)val;
    }
  }
}

extern "C" void kernel_launch(void* const* d_in, const int* in_sizes, int n_in,
                              void* d_out, int out_size, void* d_ws, size_t ws_size,
                              hipStream_t stream) {
  const float* q   = (const float*)d_in[0];
  const float* k   = (const float*)d_in[1];
  const float* v   = (const float*)d_in[2];
  // d_in[3] = causal mask (structure known statically; unused)
  const float* w_q = (const float*)d_in[4];
  const float* b_q = (const float*)d_in[5];
  const float* w_k = (const float*)d_in[6];
  const float* b_k = (const float*)d_in[7];
  const float* w_v = (const float*)d_in[8];
  const float* b_v = (const float*)d_in[9];
  const float* w_o = (const float*)d_in[10];
  const float* b_o = (const float*)d_in[11];

  const size_t np = (size_t)Bsz * Ssz * Esz;   // 4M elems per tensor
  __bf16* qb = (__bf16*)d_ws;                  // bf16 Q projection [B,S,E]
  __bf16* kb = qb + np;                        // bf16 K projection [B,S,E]
  __bf16* vt = kb + np;                        // bf16 V projection, transposed [B,H,DK,S]
  __bf16* cb = vt + np;                        // attention context (bf16) [B,S,E]

  dim3 ggrid(Esz / 128, (Bsz * Ssz) / 128);    // (8, 32)
  gemm_bias_wmma<float, 1><<<ggrid, 256, 0, stream>>>(q, w_q, b_q, qb, Bsz * Ssz, Esz, Esz);
  gemm_bias_wmma<float, 1><<<ggrid, 256, 0, stream>>>(k, w_k, b_k, kb, Bsz * Ssz, Esz, Esz);
  gemm_bias_wmma<float, 2><<<ggrid, 256, 0, stream>>>(v, w_v, b_v, vt, Bsz * Ssz, Esz, Esz);

  dim3 agrid(Ssz / 128, Bsz * Hsz);            // (16, 32)
  flash_attn_wmma<<<agrid, 256, 0, stream>>>(qb, kb, vt, cb);

  gemm_bias_wmma<__bf16, 0><<<ggrid, 256, 0, stream>>>(cb, w_o, b_o, d_out, Bsz * Ssz, Esz, Esz);
}
